// PathwayGNN_66709432042191
// MI455X (gfx1250) — matrix operand order repaired and checked by reading, not compiled, not run on hardware
//
#include <hip/hip_runtime.h>
#include <math.h>

// ---------------------------------------------------------------------------
// Problem constants (match reference)
// ---------------------------------------------------------------------------
#define NN 100000            // nodes per type
#define EE 500000            // edges per relation
#define HIDD 128
#define HEADS 4
#define HDIM 32              // HID/HEADS
#define FDIM 8
#define FFNH 512
#define NH (NN * HIDD)       // 12,800,000 elems per node-feature buffer

typedef __attribute__((ext_vector_type(16))) __bf16 v16bf;
typedef __attribute__((ext_vector_type(8)))  float  v8f;

// ---------------------------------------------------------------------------
// Workspace layout (bytes)
// ---------------------------------------------------------------------------
static const size_t OFF_HNB   = 0;                                   // 3 * NH bf16
static const size_t OFF_ACC   = OFF_HNB  + (size_t)3 * NH * 2;       // 3 * NH f32
static const size_t OFF_POOLA = OFF_ACC  + (size_t)3 * NH * 4;       // NH f32
static const size_t OFF_POOLB = OFF_POOLA + (size_t)NH * 4;          // NH f32
static const size_t OFF_POOLC = OFF_POOLB + (size_t)NH * 4;          // NH f32
static const size_t OFF_POOLD = OFF_POOLC + (size_t)NH * 4;          // NH f32
static const size_t OFF_SCORE = OFF_POOLD + (size_t)NH * 4;          // EE*4 f32
static const size_t OFF_MX    = OFF_SCORE + (size_t)EE * 4 * 4;      // NN*4 f32
static const size_t OFF_DEN   = OFF_MX    + (size_t)NN * 4 * 4;      // NN*4 f32
static const size_t OFF_DEG   = OFF_DEN   + (size_t)NN * 4 * 4;      // NN f32
static const size_t OFF_WBF   = OFF_DEG   + (size_t)NN * 4;          // bf16 weights (transposed)

// element offsets inside the bf16 weight arena (all weights stored TRANSPOSED:
// WT[n][k], n = output col, k contiguous -> B-fragment = 2 contiguous b128/lane)
static const size_t W_PHYS = 0;                 // 3 * 128*128
static const size_t W_Q    = W_PHYS + 3*16384;  // 5 * 128*128
static const size_t W_K    = W_Q    + 5*16384;
static const size_t W_V    = W_K    + 5*16384;
static const size_t W_O    = W_V    + 5*16384;
static const size_t W_FF1  = W_O    + 5*16384;  // 3 * (512x128 transposed)
static const size_t W_FF2  = W_FF1  + 3*65536;  // 3 * (128x512 transposed)

// ---------------------------------------------------------------------------
// Small utility kernels
// ---------------------------------------------------------------------------
__global__ void zero_f32_kernel(float* p, int n) {
    int i = blockIdx.x * blockDim.x + threadIdx.x;
    if (i < n) p[i] = 0.0f;
}

__global__ void fill_f32_kernel(float* p, int n, float v) {
    int i = blockIdx.x * blockDim.x + threadIdx.x;
    if (i < n) p[i] = v;
}

__global__ void cvt_f32_to_bf16_kernel(const float* __restrict__ src,
                                       __bf16* __restrict__ dst, int n) {
    int i = blockIdx.x * blockDim.x + threadIdx.x;
    if (i < n) dst[i] = (__bf16)src[i];
}

// Convert one KxNc f32 weight matrix to bf16, stored transposed: dst[n*K + k]
__global__ void cvt_transpose_bf16_kernel(const float* __restrict__ src,
                                          __bf16* __restrict__ dst,
                                          int K, int Nc) {
    int i = blockIdx.x * blockDim.x + threadIdx.x;
    if (i >= K * Nc) return;
    int k = i / Nc, n = i - k * Nc;
    dst[(size_t)n * K + k] = (__bf16)src[i];
}

// ---------------------------------------------------------------------------
// LayerNorm (pre-LN): one wave per row of 128, output bf16
// ---------------------------------------------------------------------------
__global__ void ln_bf16_kernel(const float* __restrict__ x,
                               const float* __restrict__ g,
                               const float* __restrict__ b,
                               __bf16* __restrict__ y, int nrows) {
    int row  = (blockIdx.x * blockDim.x + threadIdx.x) >> 5;
    int lane = threadIdx.x & 31;
    if (row >= nrows) return;
    const float* rp = x + (size_t)row * HIDD;
    float v[4], s = 0.0f;
#pragma unroll
    for (int t = 0; t < 4; ++t) { v[t] = rp[t * 32 + lane]; s += v[t]; }
#pragma unroll
    for (int off = 16; off >= 1; off >>= 1) s += __shfl_xor(s, off, 32);
    float mean = s * (1.0f / HIDD);
    float var = 0.0f;
#pragma unroll
    for (int t = 0; t < 4; ++t) { float d = v[t] - mean; var += d * d; }
#pragma unroll
    for (int off = 16; off >= 1; off >>= 1) var += __shfl_xor(var, off, 32);
    float inv = rsqrtf(var * (1.0f / HIDD) + 1e-5f);
    __bf16* yp = y + (size_t)row * HIDD;
#pragma unroll
    for (int t = 0; t < 4; ++t) {
        int c = t * 32 + lane;
        yp[c] = (__bf16)(g[c] * (v[t] - mean) * inv + b[c]);
    }
}

// residual + LN2 fused: ha = h + acc*invCnt -> write ha to out (f32),
// normed = LN(ha) -> bf16
__global__ void resid_ln2_kernel(const float* __restrict__ h,
                                 const float* __restrict__ acc, float invCnt,
                                 const float* __restrict__ g,
                                 const float* __restrict__ b,
                                 float* __restrict__ outHa,
                                 __bf16* __restrict__ normed, int nrows) {
    int row  = (blockIdx.x * blockDim.x + threadIdx.x) >> 5;
    int lane = threadIdx.x & 31;
    if (row >= nrows) return;
    const float* hp = h   + (size_t)row * HIDD;
    const float* ap = acc + (size_t)row * HIDD;
    float v[4], s = 0.0f;
#pragma unroll
    for (int t = 0; t < 4; ++t) {
        int c = t * 32 + lane;
        v[t] = hp[c] + ap[c] * invCnt;
        s += v[t];
    }
    float* op = outHa + (size_t)row * HIDD;
#pragma unroll
    for (int t = 0; t < 4; ++t) op[t * 32 + lane] = v[t];
#pragma unroll
    for (int off = 16; off >= 1; off >>= 1) s += __shfl_xor(s, off, 32);
    float mean = s * (1.0f / HIDD);
    float var = 0.0f;
#pragma unroll
    for (int t = 0; t < 4; ++t) { float d = v[t] - mean; var += d * d; }
#pragma unroll
    for (int off = 16; off >= 1; off >>= 1) var += __shfl_xor(var, off, 32);
    float inv = rsqrtf(var * (1.0f / HIDD) + 1e-5f);
    __bf16* np = normed + (size_t)row * HIDD;
#pragma unroll
    for (int t = 0; t < 4; ++t) {
        int c = t * 32 + lane;
        np[c] = (__bf16)(g[c] * (v[t] - mean) * inv + b[c]);
    }
}

// ---------------------------------------------------------------------------
// WMMA GEMM: C[M,Nc] = A[M,K](bf16 row-major) * WT[Nc,K](bf16, transposed B)
// block = 256 threads = 8 waves; each wave: 32 rows x 32 cols (4 wmma tiles,
// 2 A-frags x 2 B-frags reused). grid = (ceil(M/256), Nc/32)
// Fast path (full 32-row tile in bounds) has NO bounds checks in the K-loop.
// mode 0: Cf = r + bias   mode 1: Cf += r + bias   mode 2: Cb = bf16(gelu(r+bias))
// ---------------------------------------------------------------------------
// A fragment (16x32 bf16): lane m=lane&15, half=lane>>4; elements 0..7 are
// K = half*8 + e, elements 8..15 are K = 16 + half*8 + (e-8) -> two
// contiguous 16B runs per lane.
__device__ inline v16bf load_a_frag_nc(const __bf16* __restrict__ A, int lda,
                                       int row0, int k0, int lane) {
    v16bf f;
    int m = lane & 15, half = lane >> 4;
    const __bf16* p = A + (size_t)(row0 + m) * lda + k0 + half * 8;
#pragma unroll
    for (int e = 0; e < 8; ++e) f[e] = p[e];
#pragma unroll
    for (int e = 0; e < 8; ++e) f[8 + e] = p[16 + e];
    return f;
}

__device__ inline v16bf load_a_frag_guard(const __bf16* __restrict__ A, int lda,
                                          int row0, int k0, int M, int lane) {
    v16bf f;
    int m = lane & 15, half = lane >> 4;
    int r = row0 + m;
    if (r < M) {
        const __bf16* p = A + (size_t)r * lda + k0 + half * 8;
#pragma unroll
        for (int e = 0; e < 8; ++e) f[e] = p[e];
#pragma unroll
        for (int e = 0; e < 8; ++e) f[8 + e] = p[16 + e];
    } else {
#pragma unroll
        for (int i = 0; i < 16; ++i) f[i] = (__bf16)0.0f;
    }
    return f;
}

// B fragment (32x16 bf16) from transposed weights WT[n][k]: lane n=lane&15,
// half=lane>>4; elements 0..7 are K = half*8 + e, elements 8..15 are
// K = 16 + half*8 + (e-8) -> two contiguous 16B runs per lane.
__device__ inline v16bf load_bT_frag(const __bf16* __restrict__ WT, int K,
                                     int colT, int k0, int lane) {
    v16bf f;
    int n = lane & 15, half = lane >> 4;
    const __bf16* p = WT + (size_t)(colT + n) * K + k0 + half * 8;
#pragma unroll
    for (int e = 0; e < 8; ++e) f[e] = p[e];
#pragma unroll
    for (int e = 0; e < 8; ++e) f[8 + e] = p[16 + e];
    return f;
}

__global__ void wmma_gemm_kernel(const __bf16* __restrict__ A,
                                 const __bf16* __restrict__ WT,
                                 const float* __restrict__ bias,
                                 float* __restrict__ Cf,
                                 __bf16* __restrict__ Cb,
                                 int M, int K, int Nc, int mode) {
    int wave = threadIdx.x >> 5;
    int lane = threadIdx.x & 31;
    int row0 = blockIdx.x * 256 + wave * 32;   // wave handles rows [row0,row0+32)
    int col0 = blockIdx.y * 32;                // block handles cols [col0,col0+32)

    v8f acc00 = {}, acc01 = {}, acc10 = {}, acc11 = {};
    if (row0 + 32 <= M) {
        // ---- hot path: no bounds checks, pure load+wmma stream ----
        for (int k0 = 0; k0 < K; k0 += 32) {
            v16bf a0 = load_a_frag_nc(A, K, row0,      k0, lane);
            v16bf a1 = load_a_frag_nc(A, K, row0 + 16, k0, lane);
            v16bf b0 = load_bT_frag(WT, K, col0,      k0, lane);
            v16bf b1 = load_bT_frag(WT, K, col0 + 16, k0, lane);
            acc00 = __builtin_amdgcn_wmma_f32_16x16x32_bf16(false, a0, false, b0,
                                                            (short)0, acc00, false, false);
            acc01 = __builtin_amdgcn_wmma_f32_16x16x32_bf16(false, a0, false, b1,
                                                            (short)0, acc01, false, false);
            acc10 = __builtin_amdgcn_wmma_f32_16x16x32_bf16(false, a1, false, b0,
                                                            (short)0, acc10, false, false);
            acc11 = __builtin_amdgcn_wmma_f32_16x16x32_bf16(false, a1, false, b1,
                                                            (short)0, acc11, false, false);
        }
        int n  = lane & 15;
        int mh = lane >> 4;
        float bv0 = bias ? bias[col0 + n]      : 0.0f;
        float bv1 = bias ? bias[col0 + 16 + n] : 0.0f;
#pragma unroll
        for (int rt = 0; rt < 2; ++rt) {
            const v8f& aL = rt ? acc10 : acc00;
            const v8f& aR = rt ? acc11 : acc01;
            int rbase = row0 + rt * 16;
#pragma unroll
            for (int r = 0; r < 8; ++r) {
                int row = rbase + r + 8 * mh;
                size_t i0 = (size_t)row * Nc + col0 + n;
                size_t i1 = i0 + 16;
                float v0 = aL[r] + bv0;
                float v1 = aR[r] + bv1;
                if (mode == 0) {
                    Cf[i0] = v0; Cf[i1] = v1;
                } else if (mode == 1) {
                    Cf[i0] += v0; Cf[i1] += v1;
                } else {
                    Cb[i0] = (__bf16)(0.5f * v0 * (1.0f + erff(v0 * 0.70710678f)));
                    Cb[i1] = (__bf16)(0.5f * v1 * (1.0f + erff(v1 * 0.70710678f)));
                }
            }
        }
    } else if (row0 < M) {
        // ---- edge tile: per-lane guarded loads ----
        for (int k0 = 0; k0 < K; k0 += 32) {
            v16bf a0 = load_a_frag_guard(A, K, row0,      k0, M, lane);
            v16bf a1 = load_a_frag_guard(A, K, row0 + 16, k0, M, lane);
            v16bf b0 = load_bT_frag(WT, K, col0,      k0, lane);
            v16bf b1 = load_bT_frag(WT, K, col0 + 16, k0, lane);
            acc00 = __builtin_amdgcn_wmma_f32_16x16x32_bf16(false, a0, false, b0,
                                                            (short)0, acc00, false, false);
            acc01 = __builtin_amdgcn_wmma_f32_16x16x32_bf16(false, a0, false, b1,
                                                            (short)0, acc01, false, false);
            acc10 = __builtin_amdgcn_wmma_f32_16x16x32_bf16(false, a1, false, b0,
                                                            (short)0, acc10, false, false);
            acc11 = __builtin_amdgcn_wmma_f32_16x16x32_bf16(false, a1, false, b1,
                                                            (short)0, acc11, false, false);
        }
        int n  = lane & 15;
        int mh = lane >> 4;
        float bv0 = bias ? bias[col0 + n]      : 0.0f;
        float bv1 = bias ? bias[col0 + 16 + n] : 0.0f;
#pragma unroll
        for (int rt = 0; rt < 2; ++rt) {
            const v8f& aL = rt ? acc10 : acc00;
            const v8f& aR = rt ? acc11 : acc01;
            int rbase = row0 + rt * 16;
#pragma unroll
            for (int r = 0; r < 8; ++r) {
                int row = rbase + r + 8 * mh;
                if (row >= M) continue;
                size_t i0 = (size_t)row * Nc + col0 + n;
                size_t i1 = i0 + 16;
                float v0 = aL[r] + bv0;
                float v1 = aR[r] + bv1;
                if (mode == 0) {
                    Cf[i0] = v0; Cf[i1] = v1;
                } else if (mode == 1) {
                    Cf[i0] += v0; Cf[i1] += v1;
                } else {
                    Cb[i0] = (__bf16)(0.5f * v0 * (1.0f + erff(v0 * 0.70710678f)));
                    Cb[i1] = (__bf16)(0.5f * v1 * (1.0f + erff(v1 * 0.70710678f)));
                }
            }
        }
    }
}

// ---------------------------------------------------------------------------
// Physics (KineticConv) edge kernels
// ---------------------------------------------------------------------------
__global__ void phys_edge_kernel(const float* __restrict__ X,
                                 const int* __restrict__ src,
                                 const int* __restrict__ dst,
                                 float* __restrict__ num,
                                 float* __restrict__ deg, int nE) {
    int e    = (blockIdx.x * blockDim.x + threadIdx.x) >> 5;
    int lane = threadIdx.x & 31;
    if (e >= nE) return;
    int s = src[e], d = dst[e];
    const float* xp = X + (size_t)s * HIDD;
    float* np = num + (size_t)d * HIDD;
#pragma unroll
    for (int t = 0; t < 4; ++t) {
        int c = t * 32 + lane;
        float m = fmaxf(xp[c], 0.0f);
        atomicAdd(&np[c], m / (1.0f + m));
    }
    if (lane == 0) atomicAdd(&deg[d], 1.0f);
}

__global__ void phys_final_kernel(float* __restrict__ acc,
                                  const float* __restrict__ num,
                                  const float* __restrict__ deg, int total) {
    int i = blockIdx.x * blockDim.x + threadIdx.x;
    if (i < total) acc[i] += num[i] / fmaxf(deg[i >> 7], 1.0f);
}

// ---------------------------------------------------------------------------
// GAT edge kernels
// ---------------------------------------------------------------------------
__device__ inline void atomicMaxF(float* addr, float val) {
    if (val >= 0.0f) atomicMax((int*)addr, __float_as_int(val));
    else             atomicMin((unsigned int*)addr, __float_as_uint(val));
}

__global__ void attn_score_kernel(const float* __restrict__ q,
                                  const float* __restrict__ k,
                                  const int* __restrict__ src,
                                  const int* __restrict__ dst,
                                  const float* __restrict__ ef,   // may be null
                                  const float* __restrict__ ep,   // may be null
                                  float* __restrict__ score,
                                  float* __restrict__ mx, int nE) {
    int e    = (blockIdx.x * blockDim.x + threadIdx.x) >> 5;
    int lane = threadIdx.x & 31;
    if (e >= nE) return;
    int s = src[e], d = dst[e];
    int head = lane >> 3, sub = lane & 7;
    const float* qp = q + (size_t)d * HIDD + head * HDIM + sub * 4;
    const float* kp = k + (size_t)s * HIDD + head * HDIM + sub * 4;
    float p = 0.0f;
#pragma unroll
    for (int i = 0; i < 4; ++i) p += qp[i] * kp[i];
#pragma unroll
    for (int off = 4; off >= 1; off >>= 1) p += __shfl_xor(p, off, 32);
    if (sub == 0) {
        float sc = p * 0.17677669529663687f;   // 1/sqrt(32)
        if (ef) {
            const float* fp = ef + (size_t)e * FDIM;
#pragma unroll
            for (int f = 0; f < FDIM; ++f) sc += fp[f] * ep[f * HEADS + head];
        }
        score[(size_t)e * HEADS + head] = sc;
        atomicMaxF(&mx[(size_t)d * HEADS + head], sc);
    }
}

__global__ void attn_exp_kernel(float* __restrict__ score,
                                const int* __restrict__ dst,
                                const float* __restrict__ mx,
                                float* __restrict__ den, int nE) {
    int i = blockIdx.x * blockDim.x + threadIdx.x;
    if (i >= nE * HEADS) return;
    int e = i >> 2, h = i & 3;
    int d = dst[e];
    float v = __expf(score[i] - mx[(size_t)d * HEADS + h]);
    score[i] = v;
    atomicAdd(&den[(size_t)d * HEADS + h], v);
}

__global__ void attn_agg_kernel(const float* __restrict__ score,
                                const float* __restrict__ den,
                                const float* __restrict__ v,
                                const int* __restrict__ src,
                                const int* __restrict__ dst,
                                float* __restrict__ aout, int nE) {
    int e    = (blockIdx.x * blockDim.x + threadIdx.x) >> 5;
    int lane = threadIdx.x & 31;
    if (e >= nE) return;
    int s = src[e], d = dst[e];
    const float* vp = v + (size_t)s * HIDD;
    float* op = aout + (size_t)d * HIDD;
#pragma unroll
    for (int t = 0; t < 4; ++t) {               // head == t ; dim = t*32+lane
        float alpha = score[(size_t)e * HEADS + t] / den[(size_t)d * HEADS + t];
        int c = t * 32 + lane;
        atomicAdd(&op[c], alpha * vp[c]);
    }
}

// ---------------------------------------------------------------------------
// Host-side launcher
// ---------------------------------------------------------------------------
static inline void launch_gemm(const __bf16* A, const __bf16* WT, const float* bias,
                               float* Cf, __bf16* Cb, int M, int K, int Nc,
                               int mode, hipStream_t stream) {
    dim3 grid((M + 255) / 256, Nc / 32);
    wmma_gemm_kernel<<<grid, 256, 0, stream>>>(A, WT, bias, Cf, Cb, M, K, Nc, mode);
}

extern "C" void kernel_launch(void* const* d_in, const int* in_sizes, int n_in,
                              void* d_out, int out_size, void* d_ws, size_t ws_size,
                              hipStream_t stream) {
    (void)in_sizes; (void)n_in; (void)out_size; (void)ws_size;
    const float* h[3] = {(const float*)d_in[0], (const float*)d_in[1],
                         (const float*)d_in[2]};
    const int*   esrc  = (const int*)d_in[3];
    const int*   edst  = (const int*)d_in[4];
    const float* efeat = (const float*)d_in[5];
    const float* physW = (const float*)d_in[6];
    const float* physB = (const float*)d_in[7];
    const float* Wq    = (const float*)d_in[8];
    const float* Wk    = (const float*)d_in[9];
    const float* Wv    = (const float*)d_in[10];
    const float* Wo    = (const float*)d_in[11];
    const float* eproj = (const float*)d_in[12];
    const float* ln1g  = (const float*)d_in[13];
    const float* ln1b  = (const float*)d_in[14];
    const float* ln2g  = (const float*)d_in[15];
    const float* ln2b  = (const float*)d_in[16];
    const float* W1    = (const float*)d_in[17];
    const float* b1    = (const float*)d_in[18];
    const float* W2    = (const float*)d_in[19];
    const float* b2    = (const float*)d_in[20];
    float* out = (float*)d_out;

    char* ws = (char*)d_ws;
    __bf16* hnb  = (__bf16*)(ws + OFF_HNB);
    float*  acc  = (float*)(ws + OFF_ACC);
    float*  pA   = (float*)(ws + OFF_POOLA);
    float*  pB   = (float*)(ws + OFF_POOLB);
    float*  pC   = (float*)(ws + OFF_POOLC);
    float*  pD   = (float*)(ws + OFF_POOLD);
    float*  scoreB = (float*)(ws + OFF_SCORE);
    float*  mx   = (float*)(ws + OFF_MX);
    float*  den  = (float*)(ws + OFF_DEN);
    float*  deg  = (float*)(ws + OFF_DEG);
    __bf16* wbf  = (__bf16*)(ws + OFF_WBF);

    const int BLK = 256;
    const int EDGE_BLKS = (EE * 32 + BLK - 1) / BLK;   // wave-per-edge grids
    const int ROW_BLKS  = (NN + 7) / 8;                // wave-per-row grids

    // ---- convert weights f32 -> bf16 arena, transposed --------------------
    auto cvtT = [&](const float* src, size_t offElems, int K, int Nc, int cnt) {
        int total = K * Nc;
        for (int m = 0; m < cnt; ++m)
            cvt_transpose_bf16_kernel<<<(total + BLK - 1) / BLK, BLK, 0, stream>>>(
                src + (size_t)m * total, wbf + offElems + (size_t)m * total, K, Nc);
    };
    cvtT(physW, W_PHYS, HIDD, HIDD, 3);
    cvtT(Wq, W_Q, HIDD, HIDD, 5);
    cvtT(Wk, W_K, HIDD, HIDD, 5);
    cvtT(Wv, W_V, HIDD, HIDD, 5);
    cvtT(Wo, W_O, HIDD, HIDD, 5);
    cvtT(W1, W_FF1, HIDD, FFNH, 3);
    cvtT(W2, W_FF2, FFNH, HIDD, 3);

    // ---- zero aggregation buffers ----------------------------------------
    zero_f32_kernel<<<((3 * NH) + BLK - 1) / BLK, BLK, 0, stream>>>(acc, 3 * NH);

    // ---- Pre-LN of all three node types into bf16 -------------------------
    for (int i = 0; i < 3; ++i)
        ln_bf16_kernel<<<ROW_BLKS, BLK, 0, stream>>>(
            h[i], ln1g + i * HIDD, ln1b + i * HIDD, hnb + (size_t)i * NH, NN);

    // ---- Physics relations (KineticConv) ----------------------------------
    // PHYS: (src_type, dst_type) = {(1,2),(2,1),(0,2)} at relations 0..2
    const int phys_s[3] = {1, 2, 0};
    const int phys_d[3] = {2, 1, 2};
    for (int r = 0; r < 3; ++r) {
        launch_gemm(hnb + (size_t)phys_s[r] * NH, wbf + W_PHYS + (size_t)r * 16384,
                    physB + r * HIDD, pA, nullptr, NN, HIDD, HIDD, 0, stream);
        zero_f32_kernel<<<(NH + BLK - 1) / BLK, BLK, 0, stream>>>(pB, NH);
        zero_f32_kernel<<<(NN + BLK - 1) / BLK, BLK, 0, stream>>>(deg, NN);
        phys_edge_kernel<<<EDGE_BLKS, BLK, 0, stream>>>(
            pA, esrc + (size_t)r * EE, edst + (size_t)r * EE, pB, deg, EE);
        phys_final_kernel<<<(NH + BLK - 1) / BLK, BLK, 0, stream>>>(
            acc + (size_t)phys_d[r] * NH, pB, deg, NH);
    }

    // ---- Learned relations (GAT) ------------------------------------------
    // LEARN: (src_type, dst_type, feat_idx) at relations 3..7
    const int l_s[5]  = {1, 0, 0, 1, 1};
    const int l_d[5]  = {0, 0, 2, 2, 1};
    const int l_fi[5] = {0, -1, 1, 2, 3};
    for (int j = 0; j < 5; ++j) {
        int rel = 3 + j;
        // q = hn[dst] @ Wq ; k = hn[src] @ Wk ; v = hn[src] @ Wv
        launch_gemm(hnb + (size_t)l_d[j] * NH, wbf + W_Q + (size_t)j * 16384,
                    nullptr, pA, nullptr, NN, HIDD, HIDD, 0, stream);
        launch_gemm(hnb + (size_t)l_s[j] * NH, wbf + W_K + (size_t)j * 16384,
                    nullptr, pB, nullptr, NN, HIDD, HIDD, 0, stream);
        launch_gemm(hnb + (size_t)l_s[j] * NH, wbf + W_V + (size_t)j * 16384,
                    nullptr, pC, nullptr, NN, HIDD, HIDD, 0, stream);

        zero_f32_kernel<<<(NH + BLK - 1) / BLK, BLK, 0, stream>>>(pD, NH);
        zero_f32_kernel<<<(NN * 4 + BLK - 1) / BLK, BLK, 0, stream>>>(den, NN * 4);
        fill_f32_kernel<<<(NN * 4 + BLK - 1) / BLK, BLK, 0, stream>>>(
            mx, NN * 4, -INFINITY);

        const float* ef = (l_fi[j] >= 0) ? efeat + (size_t)l_fi[j] * EE * FDIM : nullptr;
        const float* ep = (l_fi[j] >= 0) ? eproj + (size_t)l_fi[j] * FDIM * HEADS : nullptr;
        const int* sp = esrc + (size_t)rel * EE;
        const int* dp = edst + (size_t)rel * EE;

        attn_score_kernel<<<EDGE_BLKS, BLK, 0, stream>>>(pA, pB, sp, dp, ef, ep,
                                                         scoreB, mx, EE);
        attn_exp_kernel<<<(EE * 4 + BLK - 1) / BLK, BLK, 0, stream>>>(
            scoreB, dp, mx, den, EE);
        attn_agg_kernel<<<EDGE_BLKS, BLK, 0, stream>>>(scoreB, den, pC, sp, dp,
                                                       pD, EE);

        // aout -> bf16 (reuse poolC; v no longer needed), then acc += aout @ Wo
        cvt_f32_to_bf16_kernel<<<(NH + BLK - 1) / BLK, BLK, 0, stream>>>(
            pD, (__bf16*)pC, NH);
        launch_gemm((__bf16*)pC, wbf + W_O + (size_t)j * 16384, nullptr,
                    acc + (size_t)l_d[j] * NH, nullptr, NN, HIDD, HIDD, 1, stream);
    }

    // ---- Final: mean over relations, residual, LN2, FFN, residual ---------
    const float invCnt[3] = {0.5f, 0.5f, 0.25f};
    for (int i = 0; i < 3; ++i) {
        float* outHa = out + (size_t)i * NH;
        __bf16* normed = (__bf16*)pC;
        resid_ln2_kernel<<<ROW_BLKS, BLK, 0, stream>>>(
            h[i], acc + (size_t)i * NH, invCnt[i],
            ln2g + i * HIDD, ln2b + i * HIDD, outHa, normed, NN);
        // hidden = gelu(normed @ W1 + b1)  (bf16, spans poolA+poolB)
        launch_gemm(normed, wbf + W_FF1 + (size_t)i * 65536, b1 + i * FFNH,
                    nullptr, (__bf16*)pA, NN, HIDD, FFNH, 2, stream);
        // out += hidden @ W2 + b2
        launch_gemm((__bf16*)pA, wbf + W_FF2 + (size_t)i * 65536, b2 + i * HIDD,
                    outHa, nullptr, NN, FFNH, HIDD, 1, stream);
    }
}
